// DeformConv_50371376447814
// MI455X (gfx1250) — matrix operand order repaired and checked by reading, not compiled, not run on hardware
//
#include <hip/hip_runtime.h>
#include <hip/hip_bf16.h>

typedef __attribute__((ext_vector_type(16))) _Float16 v16h;
typedef __attribute__((ext_vector_type(8)))  _Float16 v8h;
typedef __attribute__((ext_vector_type(8)))  float    v8f;

#define C_IN   256
#define OC_TOT 256
#define HW     4096
#define KTAPS  9
#define KTOT   2304          // 9 * 256
#define CHUNKS 72            // KTOT / 32
#define PIXT   32            // pixels per block
#define CSTR   40            // padded LDS col stride in f16 (bank-conflict free)

// -------- weight f32 [kh,kw,OC,C] -> f16 [OC][k*C + c]  (GEMM A layout) -----
__global__ __launch_bounds__(256) void weight_to_f16(
    const float* __restrict__ w, _Float16* __restrict__ wh) {
  int i   = blockIdx.x * 256 + threadIdx.x;   // 9*256*256 = 589824 exactly
  int k   = i >> 16;
  int rem = i & 65535;
  int oc  = rem >> 8;
  int c   = rem & 255;
  wh[(size_t)oc * KTOT + k * C_IN + c] = (_Float16)w[i];
}

// -------- fused deformable-conv: bilinear im2col (f32) -> f16 -> WMMA -------
__global__ __launch_bounds__(256) void deform_conv_wmma(
    const float* __restrict__ x, const float* __restrict__ off,
    const _Float16* __restrict__ wh, float* __restrict__ out) {

  __shared__ float4 s_w[KTAPS * PIXT];                 // 4 corner weights
  __shared__ int4   s_a[KTAPS * PIXT];                 // 4 corner plane offsets
  __shared__ __align__(16) _Float16 s_cols[2][PIXT * CSTR];  // double-buffered B tile

  const int tid  = threadIdx.x;
  const int lane = tid & 31;
  const int wid  = tid >> 5;
  const int pixG = blockIdx.x * PIXT;
  const int b    = pixG >> 12;          // HW = 4096 pixels per image
  const int pix0 = pixG & (HW - 1);

  // ---- precompute bilinear corners: 9 taps x 32 pixels ----
  for (int e = tid; e < KTAPS * PIXT; e += 256) {
    const int k  = e >> 5;
    const int p  = e & 31;
    const int ky = k / 3;
    const int kx = k - 3 * ky;
    const int pix = pix0 + p;
    const int ho = pix >> 6;
    const int wo = pix & 63;
    const float dy = off[(size_t)(b * 18 + 2 * k)     * HW + pix];
    const float dx = off[(size_t)(b * 18 + 2 * k + 1) * HW + pix];
    const float fy = (float)(ho - 1 + ky) + dy;   // stride 1, pad 1, dil 1
    const float fx = (float)(wo - 1 + kx) + dx;
    const float y0f = floorf(fy), x0f = floorf(fx);
    const float wy1 = fy - y0f,  wx1 = fx - x0f;
    const float wy0 = 1.0f - wy1, wx0 = 1.0f - wx1;
    const int iy0 = (int)y0f, ix0 = (int)x0f;
    const int iy1 = iy0 + 1,  ix1 = ix0 + 1;
    const bool vy0 = (iy0 >= 0) && (iy0 < 64);
    const bool vy1 = (iy1 >= 0) && (iy1 < 64);
    const bool vx0 = (ix0 >= 0) && (ix0 < 64);
    const bool vx1 = (ix1 >= 0) && (ix1 < 64);
    const int cy0 = min(max(iy0, 0), 63), cy1 = min(max(iy1, 0), 63);
    const int cx0 = min(max(ix0, 0), 63), cx1 = min(max(ix1, 0), 63);
    float4 wv;
    wv.x = (vy0 && vx0) ? wy0 * wx0 : 0.0f;
    wv.y = (vy0 && vx1) ? wy0 * wx1 : 0.0f;
    wv.z = (vy1 && vx0) ? wy1 * wx0 : 0.0f;
    wv.w = (vy1 && vx1) ? wy1 * wx1 : 0.0f;
    int4 av;
    av.x = cy0 * 64 + cx0;
    av.y = cy0 * 64 + cx1;
    av.z = cy1 * 64 + cx0;
    av.w = cy1 * 64 + cx1;
    s_w[e] = wv;
    s_a[e] = av;
  }
  __syncthreads();

  // fill one 32-wide K-chunk (tap k, channels c0..c0+31) for 32 pixels.
  // lane == pixel (coalesced gathers, x is L2-resident); each wave: 4 channels.
  auto fill = [&](int ch, int buf) {
    const int k  = ch >> 3;
    const int c0 = (ch & 7) << 5;
    const float4 wv = s_w[k * PIXT + lane];
    const int4   av = s_a[k * PIXT + lane];
#pragma unroll
    for (int cc = 0; cc < 4; ++cc) {
      const int c = c0 + wid * 4 + cc;
      const float* xp = x + ((size_t)(b * C_IN + c) << 12);
      const float v = wv.x * xp[av.x] + wv.y * xp[av.y] +
                      wv.z * xp[av.z] + wv.w * xp[av.w];
      s_cols[buf][lane * CSTR + (c - c0)] = (_Float16)v;
    }
  };

  v8f acc00 = {}; v8f acc01 = {}; v8f acc10 = {}; v8f acc11 = {};
  const int nn   = lane & 15;
  const int half = lane >> 4;
  const _Float16* arow0 = wh + (size_t)(wid * 32 + nn) * KTOT;   // M-tile 0 row
  const _Float16* arow1 = arow0 + (size_t)16 * KTOT;             // M-tile 1 row

  fill(0, 0);
  for (int ch = 0; ch < CHUNKS; ++ch) {
    __syncthreads();                       // fill(ch) complete / buf reusable
    if (ch + 1 < CHUNKS) fill(ch + 1, (ch + 1) & 1);

    const int kcb = ch * 32;
    // A fragment (ISA 16-bit A 16x32 layout): runs at K = half*8 and 16+half*8
    v8h a0lo = *(const v8h*)(arow0 + kcb + half * 8);
    v8h a0hi = *(const v8h*)(arow0 + kcb + 16 + half * 8);
    v8h a1lo = *(const v8h*)(arow1 + kcb + half * 8);
    v8h a1hi = *(const v8h*)(arow1 + kcb + 16 + half * 8);
    v16h a0 = __builtin_shufflevector(a0lo, a0hi, 0,1,2,3,4,5,6,7,8,9,10,11,12,13,14,15);
    v16h a1 = __builtin_shufflevector(a1lo, a1hi, 0,1,2,3,4,5,6,7,8,9,10,11,12,13,14,15);
    // B fragment: column N=nn of pixel tile, K runs at half*8 and 16+half*8
    const _Float16* cb = &s_cols[ch & 1][0];
    v8h b0lo = *(const v8h*)(cb + nn * CSTR + half * 8);
    v8h b0hi = *(const v8h*)(cb + nn * CSTR + 16 + half * 8);
    v8h b1lo = *(const v8h*)(cb + (16 + nn) * CSTR + half * 8);
    v8h b1hi = *(const v8h*)(cb + (16 + nn) * CSTR + 16 + half * 8);
    v16h b0 = __builtin_shufflevector(b0lo, b0hi, 0,1,2,3,4,5,6,7,8,9,10,11,12,13,14,15);
    v16h b1 = __builtin_shufflevector(b1lo, b1hi, 0,1,2,3,4,5,6,7,8,9,10,11,12,13,14,15);

    acc00 = __builtin_amdgcn_wmma_f32_16x16x32_f16(false, a0, false, b0, (short)0, acc00, false, false);
    acc01 = __builtin_amdgcn_wmma_f32_16x16x32_f16(false, a0, false, b1, (short)0, acc01, false, false);
    acc10 = __builtin_amdgcn_wmma_f32_16x16x32_f16(false, a1, false, b0, (short)0, acc10, false, false);
    acc11 = __builtin_amdgcn_wmma_f32_16x16x32_f16(false, a1, false, b1, (short)0, acc11, false, false);
  }

  // ---- store D tiles: C/D layout => M = r + 8*half, N = nn ----
  v8f accs[2][2] = {{acc00, acc01}, {acc10, acc11}};
#pragma unroll
  for (int m = 0; m < 2; ++m) {
#pragma unroll
    for (int nt = 0; nt < 2; ++nt) {
      const int ocb    = wid * 32 + m * 16;
      const int pixImg = pix0 + nt * 16 + nn;
      float* op = out + ((size_t)b * OC_TOT << 12) + pixImg;
#pragma unroll
      for (int r = 0; r < 8; ++r) {
        const int oc = ocb + r + 8 * half;
        op[(size_t)oc << 12] = accs[m][nt][r];
      }
    }
  }
}

extern "C" void kernel_launch(void* const* d_in, const int* in_sizes, int n_in,
                              void* d_out, int out_size, void* d_ws, size_t ws_size,
                              hipStream_t stream) {
  const float* x   = (const float*)d_in[0];
  const float* off = (const float*)d_in[1];
  const float* w   = (const float*)d_in[2];
  float* out = (float*)d_out;
  _Float16* wh = (_Float16*)d_ws;          // 256*2304*2 B = 1.18 MB scratch

  weight_to_f16<<<2304, 256, 0, stream>>>(w, wh);       // 589824 elems exactly
  deform_conv_wmma<<<1024, 256, 0, stream>>>(x, off, wh, out);  // 32768/32 pix tiles
}
